// WorldModel_90245852823590
// MI455X (gfx1250) — matrix operand order repaired and checked by reading, not compiled, not run on hardware
//
#include <hip/hip_runtime.h>
#include <hip/hip_bf16.h>
#include <math.h>

#define DEV __device__ __forceinline__

typedef __attribute__((ext_vector_type(16))) __bf16 v16bf;
typedef __attribute__((ext_vector_type(8)))  __bf16 v8bf;
typedef __attribute__((ext_vector_type(8)))  float  v8f;

// ---- problem constants ------------------------------------------------------
constexpr int Bn    = 16;
constexpr int Ln    = 2048;
constexpr int Ntok  = Bn * Ln;          // 32768 tokens
constexpr int DM    = 512;
constexpr int DI    = 1024;
constexpr int NSs   = 4;
constexpr int DCc   = 4;
constexpr int HP    = 64;
constexpr int NHh   = 16;
constexpr int DPROJ  = 2 * DI + 2 * NSs + NHh;   // 2072
constexpr int DPROJP = 2080;                     // padded to 32
constexpr int CDIM   = DI + 2 * NSs;             // 1032
constexpr int MH    = 128;
constexpr int Sdim  = 32;
constexpr int Edim  = 1024;

// ---- small math helpers -----------------------------------------------------
DEV __bf16 f2bf(float f) {
  unsigned u = __builtin_bit_cast(unsigned, f);
  unsigned r = (u + 0x7FFFu + ((u >> 16) & 1u)) >> 16;
  return __builtin_bit_cast(__bf16, (unsigned short)r);
}
DEV float softplus_f(float x) { return x > 20.f ? x : log1pf(expf(x)); }
DEV float elu_f(float x)      { return x > 0.f ? x : expm1f(x); }
DEV float silu_f(float x)     { return x / (1.f + expf(-x)); }

// ---- WMMA fragment loads (CDNA5 wave32 layouts, CDNA5 ISA §7.12.2) --------
// A (16x32 bf16): lanes 0-15 rows M=0..15 holding K=kb..kb+7 (v0-3) and
// K=kb+16..kb+23 (v4-7), kb = (lane>=16)*8.
DEV v16bf load_fragA(const __bf16* base, int ld) {
  int lane = threadIdx.x & 31;
  int r  = lane & 15;
  int kb = (lane >> 4) << 3;
  const __bf16* q = base + (size_t)r * ld + kb;
  union { v16bf v; v8bf h[2]; } u;
  u.h[0] = *(const v8bf*)(q);
  u.h[1] = *(const v8bf*)(q + 16);
  return u.v;
}
// B (32x16 bf16) loaded from transposed weights Wt (row = output col, K contig):
// lane holds col = lane&15, K = kb..kb+7 and kb+16..kb+23, kb = (lane>=16)*8.
DEV v16bf load_fragB(const __bf16* base, int ld) {
  int lane = threadIdx.x & 31;
  int col = lane & 15;
  int kb  = (lane >> 4) << 3;
  const __bf16* q = base + (size_t)col * ld + kb;
  union { v16bf v; v8bf h[2]; } u;
  u.h[0] = *(const v8bf*)(q);
  u.h[1] = *(const v8bf*)(q + 16);
  return u.v;
}
DEV v8f wmma_bf16(v16bf a, v16bf b, v8f c) {
  return __builtin_amdgcn_wmma_f32_16x16x32_bf16(false, a, false, b, (short)0, c,
                                                 false, false);
}

// epilogue store for one 16x16 tile's per-lane column (8 rows)
DEV void store_acc(v8f acc, int rowBase, int m, const float* __restrict__ bias,
                   float* __restrict__ outF, int ldoF,
                   __bf16* __restrict__ outH, int ldoH, int mode) {
  float bv = bias[m];
#pragma unroll
  for (int r = 0; r < 8; ++r) {
    float v = acc[r] + bv;
    int row = rowBase + r;
    if (mode == 1)      outH[(size_t)row * ldoH + m] = f2bf(elu_f(v));
    else if (mode == 0) outF[(size_t)row * ldoF + m] = v;
    else {
      outF[(size_t)row * ldoF + m] = v;
      outH[(size_t)row * ldoH + m] = f2bf(v);
    }
  }
}

// ---- conversion / transpose kernels ----------------------------------------
__global__ void cvt_bf16(const float* __restrict__ x, __bf16* __restrict__ y, int n) {
  int i = blockIdx.x * 256 + threadIdx.x;
  if (i < n) y[i] = f2bf(x[i]);
}

// Wt[m*K + k] = bf16(W[k*M + m]) with zero padding for m >= M (rows up to Mpad)
__global__ void wtrans(const float* __restrict__ W, __bf16* __restrict__ Wt,
                       int K, int M, int Mpad) {
  int i = blockIdx.x * 256 + threadIdx.x;
  if (i >= Mpad * K) return;
  int m = i / K, k = i - m * K;
  float v = (m < M) ? W[(size_t)k * M + m] : 0.f;
  Wt[(size_t)m * K + k] = f2bf(v);
}

// ---- generic WMMA GEMM: out(N, M) = [A1 | A2](N, K1+K2) @ Wt^T + bias ------
// Each wave computes a 32x32 output tile (2 A-frags x 2 B-frags, 4 accums).
// mode 0: f32 store; mode 1: elu -> bf16 store; mode 2: f32 + bf16 stores.
__global__ void gemm_wmma(const __bf16* __restrict__ A1, int K1,
                          const __bf16* __restrict__ A2, int ldA2, int K2,
                          const __bf16* __restrict__ Wt,
                          const float* __restrict__ bias,
                          float* __restrict__ outF, int ldoF,
                          __bf16* __restrict__ outH, int ldoH,
                          int Mpairs, int mode) {
  int wid = (blockIdx.x * blockDim.x + threadIdx.x) >> 5;
  int rowPair = wid / Mpairs;
  int colPair = wid - rowPair * Mpairs;
  int row0 = rowPair << 5;   // 32 rows
  int m0   = colPair << 5;   // 32 cols
  int Ktot = K1 + K2;
  v8f acc00 = {}, acc01 = {}, acc10 = {}, acc11 = {};
  const __bf16* wp0 = Wt + (size_t)m0 * Ktot;
  const __bf16* wp1 = Wt + (size_t)(m0 + 16) * Ktot;

  // phase 1: A1 (ld == K1)
  const __bf16* ar0 = A1 + (size_t)row0 * K1;
  const __bf16* ar1 = A1 + (size_t)(row0 + 16) * K1;
  for (int k = 0; k < K1; k += 32) {
    if (k + 32 < K1) {
      __builtin_prefetch(ar0 + k + 32, 0, 1);
      __builtin_prefetch(ar1 + k + 32, 0, 1);
    }
    v16bf a0 = load_fragA(ar0 + k, K1);
    v16bf a1 = load_fragA(ar1 + k, K1);
    v16bf b0 = load_fragB(wp0 + k, Ktot);
    v16bf b1 = load_fragB(wp1 + k, Ktot);
    acc00 = wmma_bf16(a0, b0, acc00);
    acc01 = wmma_bf16(a0, b1, acc01);
    acc10 = wmma_bf16(a1, b0, acc10);
    acc11 = wmma_bf16(a1, b1, acc11);
  }
  // phase 2: A2 (ld == ldA2), weight rows continue at column K1
  const __bf16* br0 = A2 + (size_t)row0 * ldA2;
  const __bf16* br1 = A2 + (size_t)(row0 + 16) * ldA2;
  for (int k = 0; k < K2; k += 32) {
    if (k + 32 < K2) {
      __builtin_prefetch(br0 + k + 32, 0, 1);
      __builtin_prefetch(br1 + k + 32, 0, 1);
    }
    v16bf a0 = load_fragA(br0 + k, ldA2);
    v16bf a1 = load_fragA(br1 + k, ldA2);
    v16bf b0 = load_fragB(wp0 + K1 + k, Ktot);
    v16bf b1 = load_fragB(wp1 + K1 + k, Ktot);
    acc00 = wmma_bf16(a0, b0, acc00);
    acc01 = wmma_bf16(a0, b1, acc01);
    acc10 = wmma_bf16(a1, b0, acc10);
    acc11 = wmma_bf16(a1, b1, acc11);
  }

  int lane = threadIdx.x & 31;
  int col = lane & 15;
  int rb  = (lane >> 4) << 3;
  store_acc(acc00, row0 + rb,      m0 + col,      bias, outF, ldoF, outH, ldoH, mode);
  store_acc(acc01, row0 + rb,      m0 + 16 + col, bias, outF, ldoF, outH, ldoH, mode);
  store_acc(acc10, row0 + 16 + rb, m0 + col,      bias, outF, ldoF, outH, ldoH, mode);
  store_acc(acc11, row0 + 16 + rb, m0 + 16 + col, bias, outF, ldoF, outH, ldoH, mode);
}

// ---- in-projection with split epilogue: z | xBC | softplus(dt) -------------
DEV void inproj_store(v8f acc, int rowBase, int m,
                      const float* __restrict__ in_b,
                      const float* __restrict__ dt_bias,
                      float* __restrict__ zbuf, float* __restrict__ xpbuf,
                      float* __restrict__ dtbuf) {
  if (m >= DPROJ) return;
  float bv = in_b[m];
#pragma unroll
  for (int r = 0; r < 8; ++r) {
    int row = rowBase + r;
    float v = acc[r] + bv;
    if (m < DI)
      zbuf[(size_t)row * DI + m] = v;
    else if (m < DI + CDIM)
      xpbuf[(size_t)row * CDIM + (m - DI)] = v;
    else {
      int hc = m - (DI + CDIM);
      dtbuf[(size_t)row * NHh + hc] = softplus_f(v + dt_bias[hc]);
    }
  }
}

__global__ void inproj_wmma(const __bf16* __restrict__ A,
                            const __bf16* __restrict__ Wt,
                            const float* __restrict__ in_b,
                            const float* __restrict__ dt_bias,
                            float* __restrict__ zbuf,
                            float* __restrict__ xpbuf,
                            float* __restrict__ dtbuf) {
  const int Mpairs = DPROJP / 32;  // 65
  int wid = (blockIdx.x * blockDim.x + threadIdx.x) >> 5;
  int rowPair = wid / Mpairs;
  int colPair = wid - rowPair * Mpairs;
  int row0 = rowPair << 5;
  int m0   = colPair << 5;
  v8f acc00 = {}, acc01 = {}, acc10 = {}, acc11 = {};
  const __bf16* wp0 = Wt + (size_t)m0 * 512;
  const __bf16* wp1 = Wt + (size_t)(m0 + 16) * 512;
  const __bf16* ar0 = A + (size_t)row0 * 512;
  const __bf16* ar1 = A + (size_t)(row0 + 16) * 512;
  for (int k = 0; k < 512; k += 32) {
    if (k + 32 < 512) {
      __builtin_prefetch(ar0 + k + 32, 0, 1);
      __builtin_prefetch(ar1 + k + 32, 0, 1);
    }
    v16bf a0 = load_fragA(ar0 + k, 512);
    v16bf a1 = load_fragA(ar1 + k, 512);
    v16bf b0 = load_fragB(wp0 + k, 512);
    v16bf b1 = load_fragB(wp1 + k, 512);
    acc00 = wmma_bf16(a0, b0, acc00);
    acc01 = wmma_bf16(a0, b1, acc01);
    acc10 = wmma_bf16(a1, b0, acc10);
    acc11 = wmma_bf16(a1, b1, acc11);
  }
  int lane = threadIdx.x & 31;
  int col = lane & 15;
  int rb  = (lane >> 4) << 3;
  inproj_store(acc00, row0 + rb,      m0 + col,      in_b, dt_bias, zbuf, xpbuf, dtbuf);
  inproj_store(acc01, row0 + rb,      m0 + 16 + col, in_b, dt_bias, zbuf, xpbuf, dtbuf);
  inproj_store(acc10, row0 + 16 + rb, m0 + col,      in_b, dt_bias, zbuf, xpbuf, dtbuf);
  inproj_store(acc11, row0 + 16 + rb, m0 + 16 + col, in_b, dt_bias, zbuf, xpbuf, dtbuf);
}

// ---- causal depthwise conv (DC=4) + SiLU -----------------------------------
__global__ void conv_kernel(const float* __restrict__ xp,
                            const float* __restrict__ conv_w,
                            const float* __restrict__ conv_b,
                            float* __restrict__ xa) {
  long i = (long)blockIdx.x * 256 + threadIdx.x;
  if (i >= (long)Ntok * CDIM) return;
  int c = (int)(i % CDIM);
  long row = i / CDIM;          // token index b*L + l
  int l = (int)(row % Ln);
  float acc = conv_b[c];
#pragma unroll
  for (int k = 0; k < DCc; ++k) {
    int dl = k - (DCc - 1);
    if (l + dl >= 0) acc += xp[(row + dl) * CDIM + c] * conv_w[k * CDIM + c];
  }
  xa[i] = silu_f(acc);
}

// ---- sequential SSM scan: one block per (batch, head), threads = HP --------
__global__ void scan_kernel(const float* __restrict__ xa,
                            const float* __restrict__ dtb,
                            const float* __restrict__ A_log,
                            const float* __restrict__ D_skip,
                            float* __restrict__ y) {
  int b = blockIdx.x >> 4;
  int h = blockIdx.x & 15;
  int p = threadIdx.x;           // 0..63
  float Ac = -expf(A_log[h]);
  float Dv = D_skip[h];
  float s0 = 0.f, s1 = 0.f, s2 = 0.f, s3 = 0.f;
  const float* xab  = xa  + (size_t)b * Ln * CDIM;
  const float* dtbp = dtb + (size_t)b * Ln * NHh;
  float* yb = y + (size_t)b * Ln * DI;
  for (int t = 0; t < Ln; ++t) {
    const float* xr = xab + (size_t)t * CDIM;
    float dtv = dtbp[t * NHh + h];
    float e   = expf(dtv * Ac);
    float xv  = xr[h * HP + p];
    float dtx = dtv * xv;
    float B0 = xr[DI + 0], B1 = xr[DI + 1], B2 = xr[DI + 2], B3 = xr[DI + 3];
    float C0 = xr[DI + 4], C1 = xr[DI + 5], C2 = xr[DI + 6], C3 = xr[DI + 7];
    s0 = e * s0 + dtx * B0;
    s1 = e * s1 + dtx * B1;
    s2 = e * s2 + dtx * B2;
    s3 = e * s3 + dtx * B3;
    yb[(size_t)t * DI + h * HP + p] =
        s0 * C0 + s1 * C1 + s2 * C2 + s3 * C3 + Dv * xv;
  }
}

// ---- y * silu(z), RMSNorm, -> bf16 -----------------------------------------
__global__ void gatenorm_kernel(const float* __restrict__ y,
                                const float* __restrict__ z,
                                const float* __restrict__ norm_w,
                                __bf16* __restrict__ out) {
  int row = blockIdx.x;
  int t = threadIdx.x;
  __shared__ float red[256];
  float g[4];
  float ss = 0.f;
  const float* yr = y + (size_t)row * DI;
  const float* zr = z + (size_t)row * DI;
#pragma unroll
  for (int i = 0; i < 4; ++i) {
    int c = t + i * 256;
    float gv = yr[c] * silu_f(zr[c]);
    g[i] = gv;
    ss += gv * gv;
  }
  red[t] = ss;
  __syncthreads();
  for (int s = 128; s > 0; s >>= 1) {
    if (t < s) red[t] += red[t + s];
    __syncthreads();
  }
  float rr = rsqrtf(red[0] / (float)DI + 1e-5f);
#pragma unroll
  for (int i = 0; i < 4; ++i) {
    int c = t + i * 256;
    out[(size_t)row * DI + c] = f2bf(g[i] * rr * norm_w[c]);
  }
}

// ---- heads: split mu/log_std, softplus, reparameterize ---------------------
__global__ void heads_kernel(const float* __restrict__ pp,
                             const float* __restrict__ qq,
                             const float* __restrict__ noise_p,
                             const float* __restrict__ noise_q,
                             float* __restrict__ out) {
  int idx = blockIdx.x * 256 + threadIdx.x;
  if (idx >= Ntok * Sdim) return;
  int row = idx >> 5, s = idx & 31;
  float mup = pp[(size_t)row * 64 + s];
  float lsp = pp[(size_t)row * 64 + 32 + s];
  float muq = qq[(size_t)row * 64 + s];
  float lsq = qq[(size_t)row * 64 + 32 + s];
  float stdp = softplus_f(lsp);
  float stdq = softplus_f(lsq);
  const size_t NS32 = (size_t)Ntok * Sdim;
  out[idx]            = mup;
  out[NS32 + idx]     = stdp;
  out[2 * NS32 + idx] = muq;
  out[3 * NS32 + idx] = stdq;
  out[4 * NS32 + idx] = mup + noise_p[idx] * stdp;
  out[5 * NS32 + idx] = muq + noise_q[idx] * stdq;
}

// ---- host launch ------------------------------------------------------------
extern "C" void kernel_launch(void* const* d_in, const int* in_sizes, int n_in,
                              void* d_out, int out_size, void* d_ws, size_t ws_size,
                              hipStream_t stream) {
  (void)in_sizes; (void)n_in; (void)out_size; (void)ws_size;
  const float* actions = (const float*)d_in[0];
  // d_in[1] = s0 (unused by the reference)
  const float* o_embed = (const float*)d_in[2];
  const float* noise_p = (const float*)d_in[3];
  const float* noise_q = (const float*)d_in[4];
  const float* wa_w = (const float*)d_in[5];
  const float* wa_b = (const float*)d_in[6];
  const float* in_w = (const float*)d_in[7];
  const float* in_b = (const float*)d_in[8];
  const float* conv_w = (const float*)d_in[9];
  const float* conv_b = (const float*)d_in[10];
  const float* dt_bias = (const float*)d_in[11];
  const float* A_log = (const float*)d_in[12];
  const float* D_skip = (const float*)d_in[13];
  const float* norm_w = (const float*)d_in[14];
  const float* out_w = (const float*)d_in[15];
  const float* out_b = (const float*)d_in[16];
  const float* pw1 = (const float*)d_in[17]; const float* pb1 = (const float*)d_in[18];
  const float* pw2 = (const float*)d_in[19]; const float* pb2 = (const float*)d_in[20];
  const float* pw3 = (const float*)d_in[21]; const float* pb3 = (const float*)d_in[22];
  const float* qw1 = (const float*)d_in[23]; const float* qb1 = (const float*)d_in[24];
  const float* qw2 = (const float*)d_in[25]; const float* qb2 = (const float*)d_in[26];
  const float* qw3 = (const float*)d_in[27]; const float* qb3 = (const float*)d_in[28];

  // bump allocator over workspace
  char* wsp = (char*)d_ws;
  auto alloc = [&](size_t bytes) -> void* {
    void* r = (void*)wsp;
    wsp += (bytes + 255) & ~(size_t)255;
    return r;
  };

  __bf16* act_bf = (__bf16*)alloc((size_t)Ntok * 32 * 2);
  __bf16* oe_bf  = (__bf16*)alloc((size_t)Ntok * Edim * 2);
  __bf16* ae_bf  = (__bf16*)alloc((size_t)Ntok * DM * 2);
  __bf16* hbf    = (__bf16*)alloc((size_t)Ntok * DM * 2);
  __bf16* wa_t = (__bf16*)alloc((size_t)512 * 32 * 2);
  __bf16* in_t = (__bf16*)alloc((size_t)DPROJP * 512 * 2);
  __bf16* ow_t = (__bf16*)alloc((size_t)512 * 1024 * 2);
  __bf16* p1_t = (__bf16*)alloc((size_t)128 * 512 * 2);
  __bf16* p2_t = (__bf16*)alloc((size_t)128 * 128 * 2);
  __bf16* p3_t = (__bf16*)alloc((size_t)64 * 128 * 2);
  __bf16* q1_t = (__bf16*)alloc((size_t)128 * 1536 * 2);
  __bf16* q2_t = (__bf16*)alloc((size_t)128 * 128 * 2);
  __bf16* q3_t = (__bf16*)alloc((size_t)64 * 128 * 2);
  float* zbuf = (float*)alloc((size_t)Ntok * DI * 4);
  float* xpb  = (float*)alloc((size_t)Ntok * CDIM * 4);
  float* xab  = (float*)alloc((size_t)Ntok * CDIM * 4);
  float* dtb  = (float*)alloc((size_t)Ntok * NHh * 4);

  // aliased reuse (stream-ordered, producers dead before overwrite):
  float*  ybuf  = xpb;            // y reuses pre-conv buffer (dead after conv)
  __bf16* ynorm = (__bf16*)xab;   // normed y reuses post-conv buffer (dead after scan)
  char* zreg = (char*)zbuf;       // z dead after gatenorm -> MLP temporaries
  __bf16* ph1 = (__bf16*)(zreg);
  __bf16* ph2 = (__bf16*)(zreg + ((size_t)16 << 20));
  float*  ppf = (float*) (zreg + ((size_t)32 << 20));
  __bf16* qh1 = (__bf16*)(zreg + ((size_t)48 << 20));
  __bf16* qh2 = (__bf16*)(zreg + ((size_t)64 << 20));
  float*  qqf = (float*) (zreg + ((size_t)80 << 20));

  float* hout = (float*)d_out + (size_t)6 * Ntok * Sdim;  // 7th output: h

  // weight transpose+convert to bf16 (M-major rows for contiguous B frags)
  auto wt = [&](const float* W, __bf16* Wt, int K, int M, int Mpad) {
    int n = Mpad * K;
    wtrans<<<(n + 255) / 256, 256, 0, stream>>>(W, Wt, K, M, Mpad);
  };
  wt(wa_w, wa_t, 32, 512, 512);
  wt(in_w, in_t, 512, DPROJ, DPROJP);
  wt(out_w, ow_t, 1024, 512, 512);
  wt(pw1, p1_t, 512, 128, 128);
  wt(pw2, p2_t, 128, 128, 128);
  wt(pw3, p3_t, 128, 64, 64);
  wt(qw1, q1_t, 1536, 128, 128);
  wt(qw2, q2_t, 128, 128, 128);
  wt(qw3, q3_t, 128, 64, 64);

  cvt_bf16<<<(Ntok * 32 + 255) / 256, 256, 0, stream>>>(actions, act_bf, Ntok * 32);
  cvt_bf16<<<(Ntok * Edim + 255) / 256, 256, 0, stream>>>(o_embed, oe_bf, Ntok * Edim);

  auto gemm = [&](const __bf16* A1, int K1, const __bf16* A2, int ldA2, int K2,
                  const __bf16* Wt_, const float* bias, float* oF, int ldF,
                  __bf16* oH, int ldH, int Mpairs, int mode) {
    int waves = (Ntok / 32) * Mpairs;   // 32-row pairs per wave
    gemm_wmma<<<waves / 4, 128, 0, stream>>>(A1, K1, A2, ldA2, K2, Wt_, bias,
                                             oF, ldF, oH, ldH, Mpairs, mode);
  };

  // 1. a_embed = elu(actions @ wa_w + wa_b) -> bf16
  gemm(act_bf, 32, nullptr, 0, 0, wa_t, wa_b, nullptr, 0, ae_bf, DM, DM / 32, 1);

  // 2. zxbcdt = a_embed @ in_w + in_b, split into z / xBC / softplus(dt)
  {
    int waves = (Ntok / 32) * (DPROJP / 32);
    inproj_wmma<<<waves / 4, 128, 0, stream>>>(ae_bf, in_t, in_b, dt_bias,
                                               zbuf, xpb, dtb);
  }

  // 3. causal depthwise conv + SiLU
  {
    long n = (long)Ntok * CDIM;
    conv_kernel<<<(int)((n + 255) / 256), 256, 0, stream>>>(xpb, conv_w, conv_b, xab);
  }

  // 4. sequential SSM scan over L
  scan_kernel<<<Bn * NHh, HP, 0, stream>>>(xab, dtb, A_log, D_skip, ybuf);

  // 5. y * silu(z), RMSNorm -> bf16
  gatenorm_kernel<<<Ntok, 256, 0, stream>>>(ybuf, zbuf, norm_w, ynorm);

  // 6. h = ynorm @ out_w + out_b : f32 straight into d_out + bf16 copy
  gemm(ynorm, 1024, nullptr, 0, 0, ow_t, out_b, hout, DM, hbf, DM, DM / 32, 2);

  // 7. prior MLP
  gemm(hbf, 512, nullptr, 0, 0, p1_t, pb1, nullptr, 0, ph1, MH, MH / 32, 1);
  gemm(ph1, 128, nullptr, 0, 0, p2_t, pb2, nullptr, 0, ph2, MH, MH / 32, 1);
  gemm(ph2, 128, nullptr, 0, 0, p3_t, pb3, ppf, 64, nullptr, 0, 64 / 32, 0);

  // 8. posterior MLP (dual-A: concat(h, o_embed) without materializing)
  gemm(hbf, 512, oe_bf, Edim, Edim, q1_t, qb1, nullptr, 0, qh1, MH, MH / 32, 1);
  gemm(qh1, 128, nullptr, 0, 0, q2_t, qb2, nullptr, 0, qh2, MH, MH / 32, 1);
  gemm(qh2, 128, nullptr, 0, 0, q3_t, qb3, qqf, 64, nullptr, 0, 64 / 32, 0);

  // 9. heads: mu/std/reparameterized samples
  heads_kernel<<<(Ntok * Sdim) / 256, 256, 0, stream>>>(ppf, qqf, noise_p, noise_q,
                                                        (float*)d_out);
}
